// Net_58334245814464
// MI455X (gfx1250) — matrix operand order repaired and checked by reading, not compiled, not run on hardware
//
#include <hip/hip_runtime.h>

// ---------------------------------------------------------------------------
// 2-layer LSTM (B=4096, T=512, I=5, H=32) + final linear, fully fused.
// One wave32 owns a 16-row batch tile for all 512 timesteps.
// Recurrent GEMMs use v_wmma_f32_16x16x32_f16 (f16 in, f32 accumulate).
// v2: persistent bias C-tiles (no splat movs), sigmoid 0.5x folded into
//     i/f/o weight tiles, branchless x A-tile, unroll-2 for cross-step ILP.
// ---------------------------------------------------------------------------

typedef __attribute__((ext_vector_type(16))) _Float16     v16h;
typedef __attribute__((ext_vector_type(8)))  float        v8f;
typedef __attribute__((ext_vector_type(4)))  unsigned int v4u;

#define LSTM_H 32
#define LSTM_T 512
#define LSTM_I 5
#define NTILES 8          // 4H / 16 = 128/16 gate column tiles

union V16H { v16h v; _Float16 h[16]; v4u q[2]; };

// ---- fast activations -----------------------------------------------------
__device__ __forceinline__ float rcp_f(float x) {
#if __has_builtin(__builtin_amdgcn_rcpf)
  return __builtin_amdgcn_rcpf(x);
#else
  return 1.0f / x;
#endif
}

__device__ __forceinline__ float tanh_f(float x) {
#if __has_builtin(__builtin_amdgcn_tanhf)
  return __builtin_amdgcn_tanhf(x);            // v_tanh_f32 on gfx1250
#elif __has_builtin(__builtin_amdgcn_exp2f)
  float e = __builtin_amdgcn_exp2f(2.885390082f * x);
  return 1.0f - 2.0f * rcp_f(1.0f + e);
#else
  float e = __expf(2.0f * x);
  return 1.0f - 2.0f * rcp_f(1.0f + e);
#endif
}

// ---- WMMA wrapper ---------------------------------------------------------
__device__ __forceinline__ v8f wmma_f16(v16h a, v16h b, v8f c) {
  return __builtin_amdgcn_wmma_f32_16x16x32_f16(false, a, false, b,
                                                (short)0, c, false, false);
}

__device__ __forceinline__ v8f splat8(float s) {
  v8f r;
#pragma unroll
  for (int i = 0; i < 8; ++i) r[i] = s;
  return r;
}

// Per-gate-tile input scale: i/f/o tiles get the sigmoid 0.5x folded in;
// g tiles (4,5) stay unscaled (tanh).
__device__ __forceinline__ float tile_scale(int tile) {
  return (tile == 4 || tile == 5) ? 1.0f : 0.5f;
}

// B-matrix (K=32 x N=16) layout, 16-bit: lanes 0-15 -> halves 0..7->K=0..7, 8..15->K=16..23;
// lanes 16-31 -> K=8..15, 24..31; N = lane&15. Element B[k][n] = W[n][k] (gates=h·Wᵀ).
__device__ __forceinline__ v16h load_Btile32(const float* __restrict__ W,
                                             int tile, int lane, float scale) {
  int n  = tile * 16 + (lane & 15);
  int kb = (lane & 16) ? 8 : 0;
  const float* p = W + n * LSTM_H + kb;
  V16H u;
#pragma unroll
  for (int i = 0; i < 16; ++i) u.h[i] = (_Float16)(p[i < 8 ? i : i + 8] * scale);
  return u.v;
}

// W_ih0 is [128,5]: only K=0..4 nonzero -> all K>=5 rows are exactly zero.
// (This makes the x A-tile's hi-half-wave and halves 5..15 don't-cares.)
__device__ __forceinline__ v16h load_Btile_ih0(const float* __restrict__ W,
                                               int tile, int lane, float scale) {
  int n = tile * 16 + (lane & 15);
  V16H u;
#pragma unroll
  for (int i = 0; i < 16; ++i) u.h[i] = (_Float16)0.f;
  if (!(lane & 16)) {
#pragma unroll
    for (int i = 0; i < LSTM_I; ++i) u.h[i] = (_Float16)(W[n * LSTM_I + i] * scale);
  }
  return u.v;
}

// A-matrix (M=16 x K=32) layout, 16-bit: lane<16: M=lane, halves 0..7->K=0..7,
// 8..15->K=16..23; lane>=16: M=lane-16, halves 0..7->K=8..15, 8..15->K=24..31.
// hbuf is h[m][j] row-major f16 -> two contiguous 16B segments per lane.
__device__ __forceinline__ v16h load_Atile(const _Float16* hb, int lane) {
  int m  = lane & 15;
  int kb = (lane & 16) ? 8 : 0;
  V16H u;
  u.q[0] = *(const v4u*)(hb + m * LSTM_H + kb);
  u.q[1] = *(const v4u*)(hb + m * LSTM_H + kb + 16);
  return u.v;
}

// Elementwise LSTM cell on D-layout gate tiles. Gate order i,f,g,o ->
// tiles {0,1},{2,3},{4,5},{6,7}; jt selects j=n (0) vs j=16+n (1).
// i/f/o pre-activations arrive pre-scaled by 0.5 (folded into weights).
__device__ __forceinline__ void lstm_cell(const v8f (&acc)[NTILES],
                                          v8f& cA, v8f& cB,
                                          _Float16* hb, int lane) {
  int n  = lane & 15;
  int mr = (lane & 16) ? 8 : 0;
#pragma unroll
  for (int jt = 0; jt < 2; ++jt) {
    v8f& c = jt ? cB : cA;
#pragma unroll
    for (int r = 0; r < 8; ++r) {
      float iv = 0.5f * tanh_f(acc[0 + jt][r]) + 0.5f;
      float fv = 0.5f * tanh_f(acc[2 + jt][r]) + 0.5f;
      float gv = tanh_f(acc[4 + jt][r]);
      float ov = 0.5f * tanh_f(acc[6 + jt][r]) + 0.5f;
      float cv = fv * c[r] + iv * gv;
      c[r] = cv;
      float hv = ov * tanh_f(cv);
      hb[(mr + r) * LSTM_H + (jt * 16 + n)] = (_Float16)hv;
    }
  }
}

__global__ void __launch_bounds__(32) lstm2_fused_kernel(
    const float* __restrict__ x,
    const float* __restrict__ Wih0, const float* __restrict__ Whh0,
    const float* __restrict__ bih0, const float* __restrict__ bhh0,
    const float* __restrict__ Wih1, const float* __restrict__ Whh1,
    const float* __restrict__ bih1, const float* __restrict__ bhh1,
    const float* __restrict__ Wout, const float* __restrict__ bout,
    float* __restrict__ out)
{
  __shared__ _Float16 hbuf[16 * LSTM_H] __attribute__((aligned(16)));
  const int lane = threadIdx.x;
  const int b0   = blockIdx.x * 16;

  // ---- one-time: weights (scaled) + fused biases as persistent C tiles ----
  v16h BWih0[NTILES], BWhh0[NTILES], BWih1[NTILES], BWhh1[NTILES];
  v8f  biasC0[NTILES], biasC1[NTILES];
#pragma unroll
  for (int t = 0; t < NTILES; ++t) {
    int   n  = t * 16 + (lane & 15);
    float sc = tile_scale(t);
    BWih0[t] = load_Btile_ih0(Wih0, t, lane, sc);
    BWhh0[t] = load_Btile32(Whh0, t, lane, sc);
    BWih1[t] = load_Btile32(Wih1, t, lane, sc);
    BWhh1[t] = load_Btile32(Whh1, t, lane, sc);
    biasC0[t] = splat8((bih0[n] + bhh0[n]) * sc);
    biasC1[t] = splat8((bih1[n] + bhh1[n]) * sc);
  }

  v16h h0A, h1A;
  V16H xa;
#pragma unroll
  for (int i = 0; i < 16; ++i) xa.h[i] = (_Float16)0.f;
  h0A = xa.v;
  h1A = xa.v;
  v8f c0a = splat8(0.f), c0b = splat8(0.f);
  v8f c1a = splat8(0.f), c1b = splat8(0.f);

  // All lanes read the same 16 batch rows (hi half-wave duplicates lo rows);
  // the duplicated values land in A K-slots whose Wih0 B rows are zero.
  const float* xrow = x + (size_t)(b0 + (lane & 15)) * (LSTM_T * LSTM_I);

#pragma unroll 2
  for (int t = 0; t < LSTM_T; ++t) {
    // x_t: branchless, only halves 0..4 rewritten (5..15 stay zero forever).
    const float* xp = xrow + t * LSTM_I;
    float xv[LSTM_I];
#pragma unroll
    for (int i = 0; i < LSTM_I; ++i) xv[i] = xp[i];
    __builtin_prefetch(xp + 24 * LSTM_I, 0, 0);  // global_prefetch_b8
#pragma unroll
    for (int i = 0; i < LSTM_I; ++i) xa.h[i] = (_Float16)xv[i];

    v8f acc[NTILES];

    // ---- layer 0: gates = bias + x·Wih0ᵀ + h0·Whh0ᵀ ----
#pragma unroll
    for (int tt = 0; tt < NTILES; ++tt) {
      v8f c = wmma_f16(xa.v, BWih0[tt], biasC0[tt]);
      acc[tt] = wmma_f16(h0A, BWhh0[tt], c);
    }
    lstm_cell(acc, c0a, c0b, hbuf, lane);
    h0A = load_Atile(hbuf, lane);          // D-layout -> A-layout via LDS

    // ---- layer 1: gates = bias + h0·Wih1ᵀ + h1·Whh1ᵀ ----
#pragma unroll
    for (int tt = 0; tt < NTILES; ++tt) {
      v8f c = wmma_f16(h0A, BWih1[tt], biasC1[tt]);
      acc[tt] = wmma_f16(h1A, BWhh1[tt], c);
    }
    lstm_cell(acc, c1a, c1b, hbuf, lane);
    h1A = load_Atile(hbuf, lane);
  }

  // ---- epilogue: out[m] = h1_final[m,:]·Wout + bout ----
  // h1A lane<16 holds K in {0..7,16..23}; lane>=16 holds {8..15,24..31}.
  V16H ua; ua.v = h1A;
  int kb = (lane & 16) ? 8 : 0;
  float partial = 0.f;
#pragma unroll
  for (int i = 0; i < 16; ++i) {
    int k = (i & 7) + kb + ((i >> 3) ? 16 : 0);
    partial += (float)ua.h[i] * Wout[k];
  }
  partial += __shfl_xor(partial, 16, 32);   // combine the two K half-waves
  if (lane < 16) out[b0 + lane] = partial + bout[0];
}

extern "C" void kernel_launch(void* const* d_in, const int* in_sizes, int n_in,
                              void* d_out, int out_size, void* d_ws, size_t ws_size,
                              hipStream_t stream) {
  const float* x    = (const float*)d_in[0];
  const float* Wih0 = (const float*)d_in[1];
  const float* Whh0 = (const float*)d_in[2];
  const float* bih0 = (const float*)d_in[3];
  const float* bhh0 = (const float*)d_in[4];
  const float* Wih1 = (const float*)d_in[5];
  const float* Whh1 = (const float*)d_in[6];
  const float* bih1 = (const float*)d_in[7];
  const float* bhh1 = (const float*)d_in[8];
  const float* Wout = (const float*)d_in[9];
  const float* bout = (const float*)d_in[10];
  float* out = (float*)d_out;

  int B = in_sizes[0] / (LSTM_T * LSTM_I);   // 4096
  int nblk = B / 16;                         // 256 single-wave workgroups
  lstm2_fused_kernel<<<nblk, 32, 0, stream>>>(x, Wih0, Whh0, bih0, bhh0,
                                              Wih1, Whh1, bih1, bhh1,
                                              Wout, bout, out);
}
